// GAT2_LSTM_Model_52604759441755
// MI455X (gfx1250) — compile-verified, matrix operand order checked by reading, not run on hardware
//

#include <hip/hip_runtime.h>
#include <hip/hip_bf16.h>

// ---------------- problem constants ----------------
#define kB   32
#define kT   16
#define kN   128
#define kC   32
#define kH   4
#define kF1  8
#define kF2  32
#define kGH  32
#define kHF1 32     // H*F1
#define kHF2 128    // H*F2
#define kD2  16384  // N*H*F2
#define kDP  4096   // N*GH
#define kLH  512
#define kOUT 128
#define kDEG 16
#define kEPS 1e-5f

// ---------------- WMMA types ----------------
typedef __attribute__((ext_vector_type(16))) __bf16 v16bf;
typedef __attribute__((ext_vector_type(2)))  __bf16 v2bf;
typedef __attribute__((ext_vector_type(8)))  float  v8f;
typedef __attribute__((ext_vector_type(4)))  float  f4;

union BfVec { v16bf v; unsigned short u[16]; };

// RNE fp32->bf16 pair, packed with a single v_perm_b32:
// r = u + 0x7FFF + lsb(u>>16); result dword = {rb[31:16], ra[31:16]}
__device__ __forceinline__ unsigned cvt2bf(float a, float b) {
#if __has_builtin(__builtin_amdgcn_cvt_pk_bf16_f32)
  union { v2bf v; unsigned u; } r;
  r.v = __builtin_amdgcn_cvt_pk_bf16_f32(a, b);
  return r.u;
#else
  unsigned ua = __float_as_uint(a), ub = __float_as_uint(b);
  unsigned ra = ua + 0x7FFFu + ((ua >> 16) & 1u);
  unsigned rb = ub + 0x7FFFu + ((ub >> 16) & 1u);
  // perm(src0=rb, src1=ra, sel): dst bytes {rb[3],rb[2],ra[3],ra[2]}
  return __builtin_amdgcn_perm(rb, ra, 0x07060302u);
#endif
}
__device__ __forceinline__ uint2 pack4bf(f4 v) {
  uint2 r;
  r.x = cvt2bf(v.x, v.y);
  r.y = cvt2bf(v.z, v.w);
  return r;
}

__device__ __forceinline__ float sigmf(float x) { return 1.0f / (1.0f + expf(-x)); }

// =====================================================================
// Generic  C = act( A[M,K] * B[N,K]^T  (+Cin) (+bias) )  via bf16 WMMA.
// fp32 operands streamed through double-buffered LDS with on-the-fly
// bf16 conversion. Block: 256 threads = 8 waves; macro-tile BM=128 x
// BN=64; each wave owns a 32x32 sub-tile -> 4 x v_wmma_f32_16x16x32_bf16
// per K-step. One barrier per K-step: prefetch k+1 (global->regs) |
// compute from buf[cur] | store regs->buf[cur^1] | barrier.
// K must be a multiple of 32 (all ours are); lda/ldb multiples of 4.
// GUARD=false when M%BM==0 && N%BN==0 (all heavy GEMMs).
// =====================================================================
#define BM 128
#define BN 64
#define BK 32

template<bool GUARD>
__global__ __launch_bounds__(256)
void gemm_wmma_bf16(const float* __restrict__ A, const float* __restrict__ Bw,
                    const float* __restrict__ bias, const float* __restrict__ Cin,
                    float* __restrict__ C,
                    int M, int N, int K, int lda, int ldb, int ldc,
                    long aStride, long bStride, long cStride, long cinStride,
                    int act)
{
  const int bz = blockIdx.z;
  A  += (long)bz * aStride;
  Bw += (long)bz * bStride;
  C  += (long)bz * cStride;
  if (Cin) Cin += (long)bz * cinStride;

  __shared__ unsigned short As[2][BM][BK];   // 16 KB
  __shared__ unsigned short Bs[2][BN][BK];   // 8 KB

  const int tid  = threadIdx.x;
  const int lane = tid & 31;
  const int wave = tid >> 5;        // 0..7
  const int half = lane >> 4;       // 0/1
  const int r    = lane & 15;
  const int wrow = wave >> 1;       // 0..3 -> 32-row slab
  const int wcol = wave & 1;        // 0..1 -> 32-col slab

  const int m0 = blockIdx.y * BM;
  const int n0 = blockIdx.x * BN;

  const int sRow = tid >> 3;              // 0..31 (+32*j)
  const int sC4  = (tid & 7) * 4;         // 0,4,...,28

  const int ar0 = wrow * 32 + r;
  const int ar1 = ar0 + 16;
  const int bc0 = wcol * 32 + r;
  const int bc1 = bc0 + 16;

  f4 ra[4], rb[2];
  auto loadTiles = [&](int k0) {
#pragma unroll
    for (int j = 0; j < 4; ++j) {
      int gm = m0 + sRow + 32 * j;
      ra[j] = f4{};
      if (!GUARD || gm < M) ra[j] = *(const f4*)(A + (long)gm * lda + k0 + sC4);
    }
#pragma unroll
    for (int j = 0; j < 2; ++j) {
      int gn = n0 + sRow + 32 * j;
      rb[j] = f4{};
      if (!GUARD || gn < N) rb[j] = *(const f4*)(Bw + (long)gn * ldb + k0 + sC4);
    }
  };
  auto storeTiles = [&](int buf) {
#pragma unroll
    for (int j = 0; j < 4; ++j)
      *(uint2*)&As[buf][sRow + 32 * j][sC4] = pack4bf(ra[j]);
#pragma unroll
    for (int j = 0; j < 2; ++j)
      *(uint2*)&Bs[buf][sRow + 32 * j][sC4] = pack4bf(rb[j]);
  };

  v8f acc00 = {}, acc01 = {}, acc10 = {}, acc11 = {};

  const int nSteps = K / BK;
  loadTiles(0);
  storeTiles(0);
  __syncthreads();
  int cur = 0;

  for (int s = 0; s < nSteps; ++s) {
    if (s + 1 < nSteps) loadTiles((s + 1) * BK);

    // fragments per ISA 16-bit 16x32 layout:
    // lanes 0-15: K 0..7 & 16..23 ; lanes 16-31: K 8..15 & 24..31
    BfVec a0, a1, b0, b1;
#pragma unroll
    for (int i = 0; i < 8; ++i) {
      int kb = ((i < 4) ? (2 * i) : (16 + 2 * (i - 4))) + half * 8;
      a0.u[2 * i] = As[cur][ar0][kb];  a0.u[2 * i + 1] = As[cur][ar0][kb + 1];
      a1.u[2 * i] = As[cur][ar1][kb];  a1.u[2 * i + 1] = As[cur][ar1][kb + 1];
      b0.u[2 * i] = Bs[cur][bc0][kb];  b0.u[2 * i + 1] = Bs[cur][bc0][kb + 1];
      b1.u[2 * i] = Bs[cur][bc1][kb];  b1.u[2 * i + 1] = Bs[cur][bc1][kb + 1];
    }
    acc00 = __builtin_amdgcn_wmma_f32_16x16x32_bf16(false, a0.v, false, b0.v, (short)0, acc00, false, false);
    acc01 = __builtin_amdgcn_wmma_f32_16x16x32_bf16(false, a0.v, false, b1.v, (short)0, acc01, false, false);
    acc10 = __builtin_amdgcn_wmma_f32_16x16x32_bf16(false, a1.v, false, b0.v, (short)0, acc10, false, false);
    acc11 = __builtin_amdgcn_wmma_f32_16x16x32_bf16(false, a1.v, false, b1.v, (short)0, acc11, false, false);

    if (s + 1 < nSteps) {
      storeTiles(cur ^ 1);
      __syncthreads();
      cur ^= 1;
    }
  }

  // epilogue: C/D layout -> VGPR v: lanes 0-15 M=v, lanes 16-31 M=v+8
  const int gmBase = m0 + wrow * 32;
  const int gnBase = n0 + wcol * 32;
#pragma unroll
  for (int mi = 0; mi < 2; ++mi) {
#pragma unroll
    for (int ni = 0; ni < 2; ++ni) {
      v8f acc = (mi == 0) ? ((ni == 0) ? acc00 : acc01)
                          : ((ni == 0) ? acc10 : acc11);
      int gn = gnBase + ni * 16 + r;
#pragma unroll
      for (int v = 0; v < 8; ++v) {
        int gm = gmBase + mi * 16 + v + half * 8;
        if (!GUARD || (gm < M && gn < N)) {
          float val = acc[v];
          if (Cin)  val += Cin[(long)gm * ldc + gn];
          if (bias) val += bias[gn];
          if (act == 1) val = fmaxf(val, 0.0f);
          C[(long)gm * ldc + gn] = val;
        }
      }
    }
  }
}

// ---------------- x [B,T,N,C] -> xT [T,B,N,C] ----------------
__global__ void permute_x(const float* __restrict__ x, float* __restrict__ xT)
{
  int idx = blockIdx.x * blockDim.x + threadIdx.x;
  if (idx >= kB * kT * kN * kC) return;
  int c = idx % kC;
  int n = (idx / kC) % kN;
  int t = (idx / (kC * kN)) % kT;
  int b = idx / (kC * kN * kT);
  xT[(((long)(t * kB + b) * kN + n)) * kC + c] = x[idx];
}

// ---------------- attention logit dots: al_s/al_d [T,B,N,H] ----------------
__global__ void attn_dots(const float* __restrict__ hlin,
                          const float* __restrict__ a_s, const float* __restrict__ a_d,
                          float* __restrict__ als, float* __restrict__ ald,
                          int F, int HF)
{
  int idx = blockIdx.x * blockDim.x + threadIdx.x;
  if (idx >= kT * kB * kN * kH) return;
  int h = idx % kH;
  int n = (idx / kH) % kN;
  int b = (idx / (kH * kN)) % kB;
  int t = idx / (kH * kN * kB);
  const float* hp  = hlin + ((long)((t * kB + b) * kN + n)) * HF + h * F;
  const float* asp = a_s + (t * kH + h) * F;
  const float* adp = a_d + (t * kH + h) * F;
  float s = 0.f, d = 0.f;
  for (int f = 0; f < F; ++f) { float v = hp[f]; s += v * asp[f]; d += v * adp[f]; }
  als[idx] = s;
  ald[idx] = d;
}

// ---------------- per-(t,b,n,h) softmax over the 17 incoming edges ----------------
// edge structure: edges n*DEG..n*DEG+15 have dst==n; edge N*DEG+n is self-loop.
__global__ void attn_softmax(const float* __restrict__ als, const float* __restrict__ ald,
                             const int* __restrict__ esrc, float* __restrict__ alph)
{
  int idx = blockIdx.x * blockDim.x + threadIdx.x;
  if (idx >= kT * kB * kN * kH) return;
  int h = idx % kH;
  int n = (idx / kH) % kN;
  int b = (idx / (kH * kN)) % kB;
  int t = idx / (kH * kN * kB);
  const float ad = ald[idx];
  const long base = (long)(t * kB + b) * kN;
  float lg[kDEG + 1];
  float mx = -1e30f;
#pragma unroll 1
  for (int j = 0; j < kDEG; ++j) {
    int s = esrc[n * kDEG + j];
    float l = als[(base + s) * kH + h] + ad;
    l = (l >= 0.f) ? l : 0.2f * l;             // leaky_relu 0.2
    lg[j] = l; mx = fmaxf(mx, l);
  }
  {
    float l = als[(base + n) * kH + h] + ad;   // self loop
    l = (l >= 0.f) ? l : 0.2f * l;
    lg[kDEG] = l; mx = fmaxf(mx, l);
  }
  float den = 0.f;
  for (int j = 0; j <= kDEG; ++j) { lg[j] = expf(lg[j] - mx); den += lg[j]; }
  float inv = 1.0f / den;
  long ob = (long)idx * (kDEG + 1);
  for (int j = 0; j <= kDEG; ++j) alph[ob + j] = lg[j] * inv;
}

// ---------------- message aggregation: out[t,b,n,h*F+f] = b + sum alpha*h[src] ------
__global__ void gat_agg(const float* __restrict__ hlin, const float* __restrict__ alph,
                        const int* __restrict__ esrc, const float* __restrict__ bias,
                        float* __restrict__ out, int F, int HF, int total)
{
  int idx = blockIdx.x * blockDim.x + threadIdx.x;
  if (idx >= total) return;
  int f = idx % F;
  int h = (idx / F) % kH;
  int n = (idx / (F * kH)) % kN;
  int b = (idx / (F * kH * kN)) % kB;
  int t = idx / (F * kH * kN * kB);
  const long hb = (long)(t * kB + b) * kN * HF + h * F + f;
  const long ab = ((long)(((t * kB + b) * kN + n) * kH + h)) * (kDEG + 1);
  float acc = bias[t * HF + h * F + f];
#pragma unroll 1
  for (int j = 0; j < kDEG; ++j) {
    int s = esrc[n * kDEG + j];
    acc += alph[ab + j] * hlin[hb + (long)s * HF];
  }
  acc += alph[ab + kDEG] * hlin[hb + (long)n * HF];
  out[(long)(t * kB + b) * kN * HF + (long)n * HF + h * F + f] = acc;
}

// ---------------- GraphNorm over nodes (per t,b,channel), optional residual+relu ----
__global__ void graph_norm_k(float* __restrict__ x, const float* __restrict__ res,
                             const float* __restrict__ w, const float* __restrict__ bb,
                             const float* __restrict__ al, int CH, int doResRelu)
{
  int tb = blockIdx.x;
  int ch = threadIdx.x;
  float* xp = x + (long)tb * kN * CH;
  float alpha = al[ch];
  float s = 0.f;
  for (int n = 0; n < kN; ++n) s += xp[n * CH + ch];
  float mean = s * (1.0f / kN);
  float vs = 0.f;
  for (int n = 0; n < kN; ++n) { float d = xp[n * CH + ch] - alpha * mean; vs += d * d; }
  float inv = rsqrtf(vs * (1.0f / kN) + kEPS);
  float ww = w[ch], bc = bb[ch];
  if (doResRelu) {
    const float* rp = res + (long)tb * kN * CH;
    for (int n = 0; n < kN; ++n) {
      float d = xp[n * CH + ch] - alpha * mean;
      float y = ww * d * inv + bc + rp[n * CH + ch];
      xp[n * CH + ch] = fmaxf(y, 0.0f);
    }
  } else {
    for (int n = 0; n < kN; ++n) {
      float d = xp[n * CH + ch] - alpha * mean;
      xp[n * CH + ch] = ww * d * inv + bc;
    }
  }
}

// ---------------- LSTM pointwise gates (torch order i,f,g,o) ----------------
__global__ void lstm_gate(const float* __restrict__ G, float* __restrict__ h,
                          float* __restrict__ c, float* __restrict__ hseq_t)
{
  int idx = blockIdx.x * blockDim.x + threadIdx.x;
  if (idx >= kB * kLH) return;
  int b = idx / kLH, j = idx % kLH;
  const float* g = G + (long)b * 4 * kLH;
  float i_ = g[j], f_ = g[kLH + j], gg = g[2 * kLH + j], o_ = g[3 * kLH + j];
  float cn = sigmf(f_) * c[idx] + sigmf(i_) * tanhf(gg);
  float hn = sigmf(o_) * tanhf(cn);
  c[idx] = cn; h[idx] = hn; hseq_t[idx] = hn;
}

__global__ void vec_add(const float* __restrict__ a, const float* __restrict__ b,
                        float* __restrict__ o, int n)
{ int i = blockIdx.x * blockDim.x + threadIdx.x; if (i < n) o[i] = a[i] + b[i]; }

__global__ void vec_zero(float* __restrict__ p, int n)
{ int i = blockIdx.x * blockDim.x + threadIdx.x; if (i < n) p[i] = 0.0f; }

// =====================================================================
extern "C" void kernel_launch(void* const* d_in, const int* in_sizes, int n_in,
                              void* d_out, int out_size, void* d_ws, size_t ws_size,
                              hipStream_t stream)
{
  const float* x      = (const float*)d_in[0];
  const int*   esrc   = (const int*)  d_in[1];
  const float* W1     = (const float*)d_in[3];
  const float* as1    = (const float*)d_in[4];
  const float* ad1    = (const float*)d_in[5];
  const float* b1     = (const float*)d_in[6];
  const float* W2     = (const float*)d_in[7];
  const float* as2    = (const float*)d_in[8];
  const float* ad2    = (const float*)d_in[9];
  const float* b2     = (const float*)d_in[10];
  const float* gn1w   = (const float*)d_in[11];
  const float* gn1b   = (const float*)d_in[12];
  const float* gn1a   = (const float*)d_in[13];
  const float* gn2w   = (const float*)d_in[14];
  const float* gn2b   = (const float*)d_in[15];
  const float* gn2a   = (const float*)d_in[16];
  const float* Wproj  = (const float*)d_in[17];
  const float* bproj  = (const float*)d_in[18];
  const float* Wih0   = (const float*)d_in[19];
  const float* Whh0   = (const float*)d_in[20];
  const float* bih0   = (const float*)d_in[21];
  const float* bhh0   = (const float*)d_in[22];
  const float* Wih1   = (const float*)d_in[23];
  const float* Whh1   = (const float*)d_in[24];
  const float* bih1   = (const float*)d_in[25];
  const float* bhh1   = (const float*)d_in[26];
  const float* Wout   = (const float*)d_in[27];
  const float* bout   = (const float*)d_in[28];
  float* out = (float*)d_out;

  float* ws = (float*)d_ws;
  size_t o = 0;
  float* xT   = ws + o; o += (size_t)kT * kB * kN * kC;
  float* lin  = ws + o; o += (size_t)kT * kB * kN * kHF2;
  float* als  = ws + o; o += (size_t)kT * kB * kN * kH;
  float* ald  = ws + o; o += (size_t)kT * kB * kN * kH;
  float* alph = ws + o; o += (size_t)kT * kB * kN * kH * 17;
  float* hmid = ws + o; o += (size_t)kT * kB * kN * kGH;
  float* gat  = ws + o; o += (size_t)kT * kB * kN * kHF2;
  float* proj = ws + o; o += (size_t)kT * kB * kDP;
  float* G0   = ws + o; o += (size_t)kT * kB * 4 * kLH;
  float* G1   = ws + o; o += (size_t)kT * kB * 4 * kLH;
  float* Gt   = ws + o; o += (size_t)kB * 4 * kLH;
  float* H0s  = ws + o; o += (size_t)kT * kB * kLH;
  float* H1s  = ws + o; o += (size_t)kT * kB * kLH;
  float* st   = ws + o; o += (size_t)4 * kB * kLH;
  float* h0 = st, *c0 = st + kB * kLH, *h1 = st + 2 * kB * kLH, *c1 = st + 3 * kB * kLH;
  float* cb0  = ws + o; o += 4 * kLH;
  float* cb1  = ws + o; o += 4 * kLH;
  (void)ws_size; (void)in_sizes; (void)n_in; (void)out_size;

  auto cdiv = [](int a, int b) { return (a + b - 1) / b; };
  auto gemm = [&](const float* A, const float* Bw, const float* bias, const float* Cin,
                  float* Cp, int M, int Nn, int K, int lda, int ldb, int ldc,
                  long aS, long bS, long cS, long ciS, int act, int batch) {
    dim3 g(cdiv(Nn, BN), cdiv(M, BM), batch);
    if ((M % BM) != 0 || (Nn % BN) != 0)
      gemm_wmma_bf16<true><<<g, dim3(256), 0, stream>>>(A, Bw, bias, Cin, Cp,
          M, Nn, K, lda, ldb, ldc, aS, bS, cS, ciS, act);
    else
      gemm_wmma_bf16<false><<<g, dim3(256), 0, stream>>>(A, Bw, bias, Cin, Cp,
          M, Nn, K, lda, ldb, ldc, aS, bS, cS, ciS, act);
  };

  { int n = kB * kT * kN * kC;
    permute_x<<<cdiv(n, 256), 256, 0, stream>>>(x, xT); }

  // GAT conv1 linear (M=4096,N=32,K=32, batched over T)
  gemm(xT, W1, nullptr, nullptr, lin,
       kB * kN, kHF1, kC, kC, kC, kHF1,
       (long)kB * kN * kC, (long)kHF1 * kC, (long)kB * kN * kHF1, 0, 0, kT);

  { int tot = kT * kB * kN * kH;
    attn_dots<<<cdiv(tot, 256), 256, 0, stream>>>(lin, as1, ad1, als, ald, kF1, kHF1);
    attn_softmax<<<cdiv(tot, 256), 256, 0, stream>>>(als, ald, esrc, alph);
    int tot2 = tot * kF1;
    gat_agg<<<cdiv(tot2, 256), 256, 0, stream>>>(lin, alph, esrc, b1, hmid, kF1, kHF1, tot2); }

  graph_norm_k<<<kT * kB, kGH, 0, stream>>>(hmid, xT, gn1w, gn1b, gn1a, kGH, 1);

  // GAT conv2 linear (M=4096,N=128,K=32)
  gemm(hmid, W2, nullptr, nullptr, lin,
       kB * kN, kHF2, kGH, kGH, kGH, kHF2,
       (long)kB * kN * kGH, (long)kHF2 * kGH, (long)kB * kN * kHF2, 0, 0, kT);

  { int tot = kT * kB * kN * kH;
    attn_dots<<<cdiv(tot, 256), 256, 0, stream>>>(lin, as2, ad2, als, ald, kF2, kHF2);
    attn_softmax<<<cdiv(tot, 256), 256, 0, stream>>>(als, ald, esrc, alph);
    int tot2 = tot * kF2;
    gat_agg<<<cdiv(tot2, 256), 256, 0, stream>>>(lin, alph, esrc, b2, gat, kF2, kHF2, tot2); }

  graph_norm_k<<<kT * kB, kHF2, 0, stream>>>(gat, nullptr, gn2w, gn2b, gn2a, kHF2, 0);

  // Projector (M=512,N=4096,K=16384) + relu
  gemm(gat, Wproj, bproj, nullptr, proj,
       kT * kB, kDP, kD2, kD2, kD2, kDP, 0, 0, 0, 0, 1, 1);

  vec_add<<<cdiv(4 * kLH, 256), 256, 0, stream>>>(bih0, bhh0, cb0, 4 * kLH);
  vec_add<<<cdiv(4 * kLH, 256), 256, 0, stream>>>(bih1, bhh1, cb1, 4 * kLH);
  vec_zero<<<cdiv(4 * kB * kLH, 256), 256, 0, stream>>>(st, 4 * kB * kLH);

  // LSTM0 input GEMM (M=512,N=2048,K=4096)
  gemm(proj, Wih0, cb0, nullptr, G0,
       kT * kB, 4 * kLH, kDP, kDP, kDP, 4 * kLH, 0, 0, 0, 0, 0, 1);

  for (int t = 0; t < kT; ++t) {
    gemm(h0, Whh0, nullptr, G0 + (size_t)t * kB * 4 * kLH, Gt,
         kB, 4 * kLH, kLH, kLH, kLH, 4 * kLH, 0, 0, 0, 0, 0, 1);
    lstm_gate<<<cdiv(kB * kLH, 256), 256, 0, stream>>>(Gt, h0, c0, H0s + (size_t)t * kB * kLH);
  }

  // LSTM1 input GEMM (M=512,N=2048,K=512)
  gemm(H0s, Wih1, cb1, nullptr, G1,
       kT * kB, 4 * kLH, kLH, kLH, kLH, 4 * kLH, 0, 0, 0, 0, 0, 1);

  for (int t = 0; t < kT; ++t) {
    gemm(h1, Whh1, nullptr, G1 + (size_t)t * kB * 4 * kLH, Gt,
         kB, 4 * kLH, kLH, kLH, kLH, 4 * kLH, 0, 0, 0, 0, 0, 1);
    lstm_gate<<<cdiv(kB * kLH, 256), 256, 0, stream>>>(Gt, h1, c1, H1s + (size_t)t * kB * kLH);
  }

  // Output head (M=32,N=128,K=512)
  gemm(h1, Wout, bout, nullptr, out,
       kB, kOUT, kLH, kLH, kLH, kOUT, 0, 0, 0, 0, 0, 1);
}
